// Model_3882650435840
// MI455X (gfx1250) — compile-verified, matrix operand order checked by reading, not run on hardware
//
#include <hip/hip_runtime.h>
#include <hip/hip_bf16.h>
#include <math.h>

#define TY_ 32
#define TX_ 1024
#define BB_ 64
#define IN_ 512
#define HH_ 512
#define CX_ 1024

typedef __attribute__((ext_vector_type(16))) _Float16 v16h;
typedef __attribute__((ext_vector_type(4)))  _Float16 v4h;
typedef __attribute__((ext_vector_type(8)))  float    v8f;
typedef __attribute__((ext_vector_type(4)))  float    v4f;

// ---------------------------------------------------------------------------
// Generic WMMA GEMM:  out[M,N] = act( A[M,K] @ Wt[N,K]^T (+ addend[M,N]) (+ bias[N]) )
// A, Wt fp32 in memory; fp16 tiles in LDS; f32 accumulate via
// v_wmma_f32_16x16x32_f16. Block tile = 32(M) x 64(N) x 64(K-stage),
// 256 threads = 8 waves (2x4), each wave owns one 16x16 C tile, 2 WMMAs per
// barrier window. Register double-buffering: next stage's global loads are
// issued before the current stage's WMMAs so the loadcnt wait lands after a
// full compute stage. Requires M%32==0, N%64==0, K%64==0 (true here).
// act: 0=none, 1=sigmoid, 2=tanh
// ---------------------------------------------------------------------------
template <typename OUT_T>
__global__ void __launch_bounds__(256) wmma_gemm(const float* __restrict__ A,
                                                 const float* __restrict__ Wt,
                                                 const float* __restrict__ addend,
                                                 const float* __restrict__ bias,
                                                 OUT_T* __restrict__ out,
                                                 int M, int N, int K, int act)
{
    __shared__ _Float16 At[32][72];   // 32 x 64 (+pad)
    __shared__ _Float16 Bt[64][72];   // 64 x 64 (+pad)

    const int m_tile = blockIdx.x * 32;
    const int n_tile = blockIdx.y * 64;
    const int tid  = threadIdx.x & 255;   // help the compiler: tid in [0,256)
    const int lane = tid & 31;
    const int wave = tid >> 5;            // 0..7
    const int wm   = (wave >> 2) * 16;    // 0,16
    const int wn   = (wave & 3) * 16;     // 0,16,32,48
    const int half = lane >> 4;
    const int l16  = lane & 15;

    v8f acc = {};
    v4f va[2], vb[4];

    auto load_stage = [&](int k0) {
#pragma unroll
        for (int j = 0; j < 2; ++j) {                 // A: 32 rows x 16 vec4, 2/thread
            int i = tid + j * 256, r = i >> 4, c4 = (i & 15) * 4;
            va[j] = *(const v4f*)&A[(size_t)(m_tile + r) * K + k0 + c4];
        }
#pragma unroll
        for (int j = 0; j < 4; ++j) {                 // B: 64 rows x 16 vec4, 4/thread
            int i = tid + j * 256, r = i >> 4, c4 = (i & 15) * 4;
            vb[j] = *(const v4f*)&Wt[(size_t)(n_tile + r) * K + k0 + c4];
        }
    };
    auto store_stage = [&]() {
#pragma unroll
        for (int j = 0; j < 2; ++j) {
            int i = tid + j * 256, r = i >> 4, c4 = (i & 15) * 4;
            v4h o = {(_Float16)va[j].x, (_Float16)va[j].y,
                     (_Float16)va[j].z, (_Float16)va[j].w};
            *(v4h*)&At[r][c4] = o;                    // ds_store_b64
        }
#pragma unroll
        for (int j = 0; j < 4; ++j) {
            int i = tid + j * 256, r = i >> 4, c4 = (i & 15) * 4;
            v4h o = {(_Float16)vb[j].x, (_Float16)vb[j].y,
                     (_Float16)vb[j].z, (_Float16)vb[j].w};
            *(v4h*)&Bt[r][c4] = o;
        }
    };

    const int nstage = K >> 6;
    load_stage(0);
    store_stage();
    __syncthreads();

    for (int s = 0; s < nstage; ++s) {
        const bool more = (s + 1 < nstage);
        if (more) load_stage((s + 1) * 64);           // prefetch next stage into VGPRs

        // two 16x16x32 WMMAs from LDS; fragments per ISA 7.12.2 16-bit layout
#pragma unroll
        for (int ks = 0; ks < 64; ks += 32) {
            v16h af, bf;
#pragma unroll
            for (int i = 0; i < 8; ++i) {
                int kb = ks + (i >> 2) * 16 + half * 8 + (i & 3) * 2;
                af[2 * i]     = At[wm + l16][kb];
                af[2 * i + 1] = At[wm + l16][kb + 1];
                bf[2 * i]     = Bt[wn + l16][kb];
                bf[2 * i + 1] = Bt[wn + l16][kb + 1];
            }
            acc = __builtin_amdgcn_wmma_f32_16x16x32_f16(
                      false, af, false, bf, (short)0, acc, false, false);
        }
        __syncthreads();
        if (more) {
            store_stage();
            __syncthreads();
        }
    }

    // C/D layout: lane n = lane&15, VGPR r -> m = r + 8*(lane>>4)
    const int n = n_tile + wn + l16;
#pragma unroll
    for (int r = 0; r < 8; ++r) {
        int m = m_tile + wm + r + 8 * half;
        float v = acc[r];
        if (addend) v += addend[(size_t)m * N + n];
        if (bias)   v += bias[n];
        if (act == 1)      v = 1.0f / (1.0f + __expf(-v));
        else if (act == 2) v = tanhf(v);
        out[(size_t)m * N + n] = (OUT_T)v;
    }
}

// ---------------------------------------------------------------------------
// Elementwise / attention kernels
// ---------------------------------------------------------------------------
// fp32 -> fp16 stream conversion; fully non-temporal (the fp16 copy is itself
// consumed with NT loads, so keep it out of L2 to preserve pctx16 residency)
__global__ void k_f32_to_f16_nt(const float* __restrict__ in, _Float16* __restrict__ out, long n4)
{
    long i = (long)blockIdx.x * blockDim.x + threadIdx.x;
    long stride = (long)gridDim.x * blockDim.x;
    const v4f* inv = (const v4f*)in;
    v4h* outv = (v4h*)out;
    for (; i < n4; i += stride) {
        v4f v = __builtin_nontemporal_load(inv + i);
        v4h o = {(_Float16)v.x, (_Float16)v.y, (_Float16)v.z, (_Float16)v.w};
        __builtin_nontemporal_store(o, outv + i);
    }
}

__global__ void k_copy_f32(const float* __restrict__ in, float* __restrict__ out, int n)
{
    int i = blockIdx.x * blockDim.x + threadIdx.x;
    if (i < n) out[i] = in[i];
}

// rh[b,j] = h[b,j] * gates[b, j]   (r = first half of [B,2H] gates)
__global__ void k_mul_rgate(const float* __restrict__ h, const float* __restrict__ gates,
                            float* __restrict__ rh)
{
    int i = blockIdx.x * blockDim.x + threadIdx.x;   // B*H
    int b = i >> 9, j = i & (HH_ - 1);
    rh[i] = h[i] * gates[b * (2 * HH_) + j];
}

// h_new = ym*(u*hprev + (1-u)*hcand) + (1-ym)*hprev ; u = gates[:, H+j]
__global__ void k_gru_combine(const float* __restrict__ gates, const float* __restrict__ hprev,
                              const float* __restrict__ hcand, const float* __restrict__ ymask,
                              float* __restrict__ hnew)
{
    int i = blockIdx.x * blockDim.x + threadIdx.x;   // B*H
    int b = i >> 9, j = i & (HH_ - 1);
    float u  = gates[b * (2 * HH_) + HH_ + j];
    float hp = hprev[i];
    float v  = u * hp + (1.0f - u) * hcand[i];
    float ym = ymask[b];
    hnew[i] = ym * v + (1.0f - ym) * hp;
}

__global__ void k_gru_combine_out(const float* __restrict__ gates, const float* __restrict__ hprev,
                                  const float* __restrict__ hcand, const float* __restrict__ ymask,
                                  float* __restrict__ hstate, float* __restrict__ hs,
                                  float* __restrict__ ss)
{
    int i = blockIdx.x * blockDim.x + threadIdx.x;   // B*H
    int b = i >> 9, j = i & (HH_ - 1);
    float u  = gates[b * (2 * HH_) + HH_ + j];
    float hp = hprev[i];
    float v  = u * hp + (1.0f - u) * hcand[i];
    float ym = ymask[b];
    v = ym * v + (1.0f - ym) * hp;
    hstate[i] = v; hs[i] = v; ss[i] = v;
}

// scores[tx*B+b] = xmask[tx,b] * sum_c U_att[c] * tanh(pctx16[tx,b,c] + hatt[b,c])
// single pass: 256 threads x 4 fp16 = 1024 = CTX. pctx16 reads are RT so the
// 134MB array stays resident in the 192MB L2 across all 32 steps.
__global__ void k_att_scores(const _Float16* __restrict__ pctx, const float* __restrict__ hatt,
                             const float* __restrict__ uatt, const float* __restrict__ xmask,
                             float* __restrict__ scores)
{
    __shared__ float red[256];
    int txb = blockIdx.x;            // tx*B + b
    int b = txb & (BB_ - 1);
    const v4h* prow = (const v4h*)(pctx + (long)txb * CX_);
    const v4f* hrow = (const v4f*)(hatt + (size_t)b * CX_);
    const v4f* urow = (const v4f*)uatt;
    int c = threadIdx.x;
    v4h p = prow[c];
    v4f hv = hrow[c];
    v4f uv = urow[c];
    float acc = uv.x * tanhf((float)p.x + hv.x)
              + uv.y * tanhf((float)p.y + hv.y)
              + uv.z * tanhf((float)p.z + hv.z)
              + uv.w * tanhf((float)p.w + hv.w);
    red[threadIdx.x] = acc;
    __syncthreads();
#pragma unroll
    for (int s = 128; s > 0; s >>= 1) {
        if (threadIdx.x < s) red[threadIdx.x] += red[threadIdx.x + s];
        __syncthreads();
    }
    if (threadIdx.x == 0) scores[txb] = red[0] * xmask[txb];
}

// masked softmax over tx for each b; one block per b (TX=1024, 256 thr x 4)
__global__ void k_softmax(const float* __restrict__ scores, const float* __restrict__ xmask,
                          float* __restrict__ watt)
{
    __shared__ float red[256];
    int b = blockIdx.x;
    float s[4], xm[4];
    float mx = -1e30f;
#pragma unroll
    for (int i = 0; i < 4; ++i) {
        int tx = threadIdx.x + i * 256;
        s[i]  = scores[tx * BB_ + b];
        xm[i] = xmask[tx * BB_ + b];
        mx = fmaxf(mx, s[i]);
    }
    red[threadIdx.x] = mx; __syncthreads();
    for (int st = 128; st > 0; st >>= 1) {
        if (threadIdx.x < st) red[threadIdx.x] = fmaxf(red[threadIdx.x], red[threadIdx.x + st]);
        __syncthreads();
    }
    mx = red[0]; __syncthreads();
    float e[4], sum = 0.0f;
#pragma unroll
    for (int i = 0; i < 4; ++i) { e[i] = __expf(s[i] - mx) * xm[i]; sum += e[i]; }
    red[threadIdx.x] = sum; __syncthreads();
    for (int st = 128; st > 0; st >>= 1) {
        if (threadIdx.x < st) red[threadIdx.x] += red[threadIdx.x + st];
        __syncthreads();
    }
    float inv = 1.0f / red[0];
#pragma unroll
    for (int i = 0; i < 4; ++i) {
        int tx = threadIdx.x + i * 256;
        watt[tx * BB_ + b] = e[i] * inv;
    }
}

// partial[p,b,n] = sum_{tx in chunk p} watt[tx,b] * ctx16[tx,b,n]
// ctx16 is streamed NON-TEMPORALLY so it never evicts pctx16 from L2.
// Strided stream (128KB/row) -> explicit prefetch two rows ahead.
__global__ void k_att_ctx_partial(const float* __restrict__ watt, const _Float16* __restrict__ ctx16,
                                  float* __restrict__ partial)
{
    int b = blockIdx.x;               // 64
    int p = blockIdx.y;               // 8
    int n4 = threadIdx.x;             // vec4 column index
    v4f acc = {0.0f, 0.0f, 0.0f, 0.0f};
    int tx0 = p * (TX_ / 8);
    for (int tx = tx0; tx < tx0 + TX_ / 8; ++tx) {
        int txb = tx * BB_ + b;
        __builtin_prefetch(ctx16 + ((long)txb + 2L * BB_) * CX_ + 4 * n4, 0, 0);
        float w = watt[txb];
        v4h c = __builtin_nontemporal_load((const v4h*)(ctx16 + (long)txb * CX_) + n4);
        acc.x += w * (float)c.x;
        acc.y += w * (float)c.y;
        acc.z += w * (float)c.z;
        acc.w += w * (float)c.w;
    }
    ((v4f*)(partial + (size_t)(p * BB_ + b) * CX_))[n4] = acc;
}

__global__ void k_att_reduce(const float* __restrict__ partial, float* __restrict__ atted,
                             float* __restrict__ atts_out)
{
    int i = blockIdx.x * blockDim.x + threadIdx.x;   // B*CTX/4 vec4 groups
    const v4f* pv = (const v4f*)partial;
    v4f acc = {0.0f, 0.0f, 0.0f, 0.0f};
#pragma unroll
    for (int p = 0; p < 8; ++p) {
        v4f v = pv[(size_t)p * (BB_ * CX_ / 4) + i];
        acc.x += v.x; acc.y += v.y; acc.z += v.z; acc.w += v.w;
    }
    ((v4f*)atted)[i] = acc;
    ((v4f*)atts_out)[i] = acc;
}

// ---------------------------------------------------------------------------
extern "C" void kernel_launch(void* const* d_in, const int* in_sizes, int n_in,
                              void* d_out, int out_size, void* d_ws, size_t ws_size,
                              hipStream_t stream)
{
    const float* y_emb      = (const float*)d_in[0];
    const float* context    = (const float*)d_in[1];
    const float* init_state = (const float*)d_in[2];
    const float* x_mask     = (const float*)d_in[3];
    const float* y_mask     = (const float*)d_in[4];
    const float* W          = (const float*)d_in[5];
    const float* U          = (const float*)d_in[6];
    const float* b          = (const float*)d_in[7];
    const float* Wx         = (const float*)d_in[8];
    const float* Ux         = (const float*)d_in[9];
    const float* bx         = (const float*)d_in[10];
    const float* Wc_att     = (const float*)d_in[11];
    const float* b_att      = (const float*)d_in[12];
    const float* W_comb_att = (const float*)d_in[13];
    const float* U_att      = (const float*)d_in[14];
    const float* U_nl       = (const float*)d_in[15];
    const float* b_nl       = (const float*)d_in[16];
    const float* Ux_nl      = (const float*)d_in[17];
    const float* bx_nl      = (const float*)d_in[18];
    const float* Wc         = (const float*)d_in[19];
    const float* Wcx        = (const float*)d_in[20];

    float* hs_out   = (float*)d_out;
    float* ss_out   = hs_out + (size_t)TY_ * BB_ * HH_;
    float* atts_out = ss_out + (size_t)TY_ * BB_ * HH_;

    char* wsp = (char*)d_ws;
    auto alloc = [&](size_t bytes) -> char* {
        char* p = wsp;
        wsp += (bytes + 255) & ~(size_t)255;
        return p;
    };
    _Float16* ctx16  = (_Float16*)alloc((size_t)TX_ * BB_ * CX_ * 2);
    _Float16* pctx16 = (_Float16*)alloc((size_t)TX_ * BB_ * CX_ * 2);
    float* xbuf   = (float*)alloc((size_t)TY_ * BB_ * 2 * HH_ * 4);
    float* xxbuf  = (float*)alloc((size_t)TY_ * BB_ * HH_ * 4);
    float* h      = (float*)alloc((size_t)BB_ * HH_ * 4);
    float* tmp1   = (float*)alloc((size_t)BB_ * 2 * HH_ * 4);
    float* rh     = (float*)alloc((size_t)BB_ * HH_ * 4);
    float* h1pre  = (float*)alloc((size_t)BB_ * HH_ * 4);
    float* h1     = (float*)alloc((size_t)BB_ * HH_ * 4);
    float* hatt   = (float*)alloc((size_t)BB_ * CX_ * 4);
    float* scores = (float*)alloc((size_t)TX_ * BB_ * 4);
    float* watt   = (float*)alloc((size_t)TX_ * BB_ * 4);
    float* partial= (float*)alloc((size_t)8 * BB_ * CX_ * 4);
    float* atted  = (float*)alloc((size_t)BB_ * CX_ * 4);
    float* t2a    = (float*)alloc((size_t)BB_ * 2 * HH_ * 4);
    float* tmp2   = (float*)alloc((size_t)BB_ * 2 * HH_ * 4);
    float* ucx    = (float*)alloc((size_t)BB_ * HH_ * 4);
    float* rh2    = (float*)alloc((size_t)BB_ * HH_ * 4);
    float* h2pre  = (float*)alloc((size_t)BB_ * HH_ * 4);

    // --- precompute (parallel phase) ---
    k_f32_to_f16_nt<<<4096, 256, 0, stream>>>(context, ctx16, (long)TX_ * BB_ * CX_ / 4);
    // pctx = context @ Wc_att^T + b_att   (stored fp16; L2-resident across scan)
    wmma_gemm<_Float16><<<dim3(TX_ * BB_ / 32, CX_ / 64), 256, 0, stream>>>(
        context, Wc_att, nullptr, b_att, pctx16, TX_ * BB_, CX_, CX_, 0);
    // x  = y_emb @ W^T + b ; xx = y_emb @ Wx^T + bx
    wmma_gemm<float><<<dim3(TY_ * BB_ / 32, 2 * HH_ / 64), 256, 0, stream>>>(
        y_emb, W, nullptr, b, xbuf, TY_ * BB_, 2 * HH_, IN_, 0);
    wmma_gemm<float><<<dim3(TY_ * BB_ / 32, HH_ / 64), 256, 0, stream>>>(
        y_emb, Wx, nullptr, bx, xxbuf, TY_ * BB_, HH_, IN_, 0);
    k_copy_f32<<<(BB_ * HH_ + 255) / 256, 256, 0, stream>>>(init_state, h, BB_ * HH_);

    const int EW = BB_ * HH_ / 256;   // elementwise grid

    // --- sequential scan ---
    for (int t = 0; t < TY_; ++t) {
        const float* xt  = xbuf  + (size_t)t * BB_ * 2 * HH_;
        const float* xxt = xxbuf + (size_t)t * BB_ * HH_;
        const float* ymt = y_mask + t * BB_;

        // GRU sub-cell 1
        wmma_gemm<float><<<dim3(BB_ / 32, 2 * HH_ / 64), 256, 0, stream>>>(
            h, U, xt, nullptr, tmp1, BB_, 2 * HH_, HH_, 1);               // sigmoid
        k_mul_rgate<<<EW, 256, 0, stream>>>(h, tmp1, rh);
        wmma_gemm<float><<<dim3(BB_ / 32, HH_ / 64), 256, 0, stream>>>(
            rh, Ux, xxt, nullptr, h1pre, BB_, HH_, HH_, 2);               // tanh
        k_gru_combine<<<EW, 256, 0, stream>>>(tmp1, h, h1pre, ymt, h1);

        // attention
        wmma_gemm<float><<<dim3(BB_ / 32, CX_ / 64), 256, 0, stream>>>(
            h1, W_comb_att, nullptr, nullptr, hatt, BB_, CX_, HH_, 0);
        k_att_scores<<<TX_ * BB_, 256, 0, stream>>>(pctx16, hatt, U_att, x_mask, scores);
        k_softmax<<<BB_, 256, 0, stream>>>(scores, x_mask, watt);
        k_att_ctx_partial<<<dim3(BB_, 8), 256, 0, stream>>>(watt, ctx16, partial);
        k_att_reduce<<<BB_ * CX_ / 4 / 256, 256, 0, stream>>>(partial, atted,
                                                              atts_out + (size_t)t * BB_ * CX_);

        // GRU sub-cell 2
        wmma_gemm<float><<<dim3(BB_ / 32, 2 * HH_ / 64), 256, 0, stream>>>(
            atted, Wc, nullptr, b_nl, t2a, BB_, 2 * HH_, CX_, 0);
        wmma_gemm<float><<<dim3(BB_ / 32, 2 * HH_ / 64), 256, 0, stream>>>(
            h1, U_nl, t2a, nullptr, tmp2, BB_, 2 * HH_, HH_, 1);          // sigmoid
        wmma_gemm<float><<<dim3(BB_ / 32, HH_ / 64), 256, 0, stream>>>(
            atted, Wcx, nullptr, bx_nl, ucx, BB_, HH_, CX_, 0);
        k_mul_rgate<<<EW, 256, 0, stream>>>(h1, tmp2, rh2);
        wmma_gemm<float><<<dim3(BB_ / 32, HH_ / 64), 256, 0, stream>>>(
            rh2, Ux_nl, ucx, nullptr, h2pre, BB_, HH_, HH_, 2);           // tanh
        k_gru_combine_out<<<EW, 256, 0, stream>>>(tmp2, h1, h2pre, ymt, h,
                                                  hs_out + (size_t)t * BB_ * HH_,
                                                  ss_out + (size_t)t * BB_ * HH_);
    }
}
